// Model_5557687681833
// MI455X (gfx1250) — matrix operand order repaired and checked by reading, not compiled, not run on hardware
//
#include <hip/hip_runtime.h>
#include <hip/hip_bf16.h>

typedef __attribute__((ext_vector_type(16))) _Float16 v16h;
typedef __attribute__((ext_vector_type(8)))  _Float16 v8h;
typedef __attribute__((ext_vector_type(8)))  float    v8f;

#define GN 100000
#define GE 400000
#define GQ 4096

// ---------------- elementwise / prep kernels ----------------

__global__ void k_g(const float* __restrict__ t, float* __restrict__ g, int e) {
  int i = blockIdx.x * blockDim.x + threadIdx.x;
  if (i < e) {
    float tl = t[e - 1], t0 = t[0];
    g[i] = (tl - t[i]) / (1.0f + tl - t0);
  }
}

// pad+convert a (rows, cols) row-major f32 matrix into (rp, kp) f16, zero filled
__global__ void k_pad_f16(const float* __restrict__ src, int rows, int cols,
                          _Float16* __restrict__ dst, int rp, int kp) {
  int i = blockIdx.x * blockDim.x + threadIdx.x;
  if (i < rp * kp) {
    int r = i / kp, c = i - r * kp;
    float v = (r < rows && c < cols) ? src[(size_t)r * cols + c] : 0.f;
    dst[i] = (_Float16)v;
  }
}

// convert n rows of f32 (ldsrc, logical width w) into f16 (lddst), zero pad cols >= w
__global__ void k_f32_to_f16_pad(const float* __restrict__ src, int ldsrc, int w,
                                 _Float16* __restrict__ dst, int lddst, int n) {
  long long i = (long long)blockIdx.x * blockDim.x + threadIdx.x;
  long long total = (long long)n * lddst;
  if (i < total) {
    int c = (int)(i % lddst);
    long long r = i / lddst;
    float v = (c < w) ? src[r * (long long)ldsrc + c] : 0.f;
    dst[i] = (_Float16)v;
  }
}

// ---------------- edge scatter (atomics resolve in L2) ----------------

__global__ void k_scatter(const int* __restrict__ u, const int* __restrict__ v,
                          const float* __restrict__ g,
                          const float* __restrict__ h, int ldh, int w,
                          float* __restrict__ agg, int lda, int e) {
  int tid = blockIdx.x * blockDim.x + threadIdx.x;
  int ed = tid >> 5, lane = tid & 31;
  if (ed >= e) return;
  int uu = u[ed], vv = v[ed];
  float ge = g[ed];
  const float* hu = h + (size_t)uu * ldh;
  const float* hv = h + (size_t)vv * ldh;
  float* au = agg + (size_t)uu * lda;
  float* av = agg + (size_t)vv * lda;
  for (int c = lane; c < w; c += 32) {
    atomicAdd(av + c, hu[c]);
    atomicAdd(au + c, hv[c]);
  }
  if (lane == 0) {
    atomicAdd(av + w, ge);
    atomicAdd(au + w, ge);
  }
}

// ---------------- batchnorm (column stats over N rows) ----------------

__global__ void k_bnstats(const float* __restrict__ a, int ld, int n,
                          float* __restrict__ sum, float* __restrict__ sq) {
  int c = threadIdx.x;                 // blockDim.x == ld
  int r0 = blockIdx.x * 256;
  int rend = r0 + 256; if (rend > n) rend = n;
  float s = 0.f, q = 0.f;
  for (int r = r0; r < rend; ++r) {    // each iteration: block reads one full row (coalesced)
    float x = a[(size_t)r * ld + c];
    s += x; q += x * x;
  }
  atomicAdd(sum + c, s);
  atomicAdd(sq + c, q);
}

__global__ void k_bnfinal(const float* __restrict__ sum, const float* __restrict__ sq,
                          float* __restrict__ mean, float* __restrict__ inv, int n) {
  int c = threadIdx.x;
  float m = sum[c] / (float)n;
  float var = sq[c] / (float)n - m * m;
  mean[c] = m;
  inv[c] = rsqrtf(var + 1e-5f);
}

__global__ void k_bnnorm(const float* __restrict__ a, const float* __restrict__ mean,
                         const float* __restrict__ inv, _Float16* __restrict__ out,
                         int ld, int n) {
  long long i = (long long)blockIdx.x * blockDim.x + threadIdx.x;
  long long total = (long long)n * ld;
  if (i < total) {
    int c = (int)(i % ld);
    out[i] = (_Float16)((a[i] - mean[c]) * inv[c]);
  }
}

// ---------------- WMMA GEMM: C(M, 16*NT slab) = A(M,kp) @ B(.,kp)^T ----------------
// One wave computes a 16-row x (16*NT)-col slab. Branch-free inner loop:
// host guarantees every launched slab is fully inside [0, jpad).
// M % 16 == 0, kp % 32 == 0, strides % 16 (f16 elems) == 0.

union AFrag { v16h v; struct { v8h lo, hi; } h; };

template <int NT, bool RELU, bool F16OUT>
__global__ __launch_bounds__(32) void k_gemm_wmma(
    const _Float16* __restrict__ A, int lda,
    const _Float16* __restrict__ B, int ldb,
    int kp, float* __restrict__ Cf, _Float16* __restrict__ Ch,
    int ldc, int jbase0) {
  int lane = threadIdx.x;              // block = 1 wave
  int m0 = blockIdx.x * 16;
  int jbase = jbase0 + blockIdx.y * (16 * NT);

  const int l15 = lane & 15;
  const int half8 = (lane >> 4) * 8;     // unused (A uses half16, like B)
  const int half16 = (lane >> 4) * 16;   // B: K sub-offset per lane-half

  v8f acc[NT];
#pragma unroll
  for (int t = 0; t < NT; ++t)
#pragma unroll
    for (int i = 0; i < 8; ++i) acc[t][i] = 0.f;

  const _Float16* arow = A + (size_t)(m0 + l15) * lda;
  const _Float16* brow[NT];
#pragma unroll
  for (int t = 0; t < NT; ++t)
    brow[t] = B + (size_t)(jbase + t * 16 + l15) * ldb + half16;

  for (int k0 = 0; k0 < kp; k0 += 32) {
    AFrag a;
    a.h.lo = *(const v8h*)(arow + k0 + half16);       // K = k0+half16 .. +7
    a.h.hi = *(const v8h*)(arow + k0 + half16 + 8);   // K = k0+half16+8 .. +15
#pragma unroll
    for (int t = 0; t < NT; ++t) {
      v16h b = *(const v16h*)(brow[t] + k0);
      acc[t] = __builtin_amdgcn_wmma_f32_16x16x32_f16(
          false, a.v, false, b, (short)0, acc[t], false, false);
    }
  }

  const int rowadd = (lane >> 4) * 8;
#pragma unroll
  for (int t = 0; t < NT; ++t) {
    int j0 = jbase + t * 16;
#pragma unroll
    for (int r = 0; r < 8; ++r) {
      int row = m0 + rowadd + r;
      int col = j0 + l15;
      float val = acc[t][r];
      if (RELU) val = fmaxf(val, 0.f);
      if (F16OUT) Ch[(size_t)row * ldc + col] = (_Float16)val;
      else        Cf[(size_t)row * ldc + col] = val;
    }
  }
}

// ---------------- heads ----------------

__global__ void k_gather_hq(const float* __restrict__ h2,
                            const int* __restrict__ qu, const int* __restrict__ qv,
                            _Float16* __restrict__ dst) {
  int i = blockIdx.x * blockDim.x + threadIdx.x;
  const int ld = 1056;
  if (i < GQ * ld) {
    int q = i / ld, c = i - q * ld;
    float v = 0.f;
    if (c < 515)        v = h2[(size_t)qu[q] * 528 + c];
    else if (c < 1030)  v = h2[(size_t)qv[q] * 528 + (c - 515)];
    dst[i] = (_Float16)v;
  }
}

__global__ void k_node_out(const float* __restrict__ y,     // (N,128) already relu'd
                           const float* __restrict__ wo,    // (10,128)
                           float* __restrict__ out, int n) {
  int i = blockIdx.x * blockDim.x + threadIdx.x;
  if (i < n * 10) {
    int r = i / 10, c = i - r * 10;
    const float* yr = y + (size_t)r * 128;
    const float* wr = wo + c * 128;
    float s = 0.f;
    for (int k = 0; k < 128; ++k) s += yr[k] * wr[k];
    out[i] = s;
  }
}

__global__ void k_link_out(const float* __restrict__ y,     // (Q,128) already relu'd
                           const float* __restrict__ wo,    // (128,)
                           float* __restrict__ out) {
  int q = blockIdx.x * blockDim.x + threadIdx.x;
  if (q < GQ) {
    const float* yr = y + (size_t)q * 128;
    float s = 0.f;
    for (int k = 0; k < 128; ++k) s += yr[k] * wo[k];
    out[q] = s;
  }
}

// ---------------- host-side GEMM dispatch ----------------

static void gemm_dispatch(const _Float16* A, int lda, const _Float16* B, int ldb,
                          int jpad, int kp, float* Cf, _Float16* Ch, int ldc,
                          bool relu, int M, hipStream_t stream) {
  bool f16out = (Ch != nullptr);
  int full = jpad / 64;            // NT=4 slabs
  int rem = (jpad % 64) / 16;      // all our jpads are 64k or 64k+16 -> rem in {0,1}
  if (full) {
    dim3 grid(M / 16, full);
    if (f16out) {
      if (relu) k_gemm_wmma<4, true,  true ><<<grid, 32, 0, stream>>>(A, lda, B, ldb, kp, Cf, Ch, ldc, 0);
      else      k_gemm_wmma<4, false, true ><<<grid, 32, 0, stream>>>(A, lda, B, ldb, kp, Cf, Ch, ldc, 0);
    } else {
      if (relu) k_gemm_wmma<4, true,  false><<<grid, 32, 0, stream>>>(A, lda, B, ldb, kp, Cf, Ch, ldc, 0);
      else      k_gemm_wmma<4, false, false><<<grid, 32, 0, stream>>>(A, lda, B, ldb, kp, Cf, Ch, ldc, 0);
    }
  }
  if (rem) {
    dim3 grid(M / 16, 1);
    int jb = full * 64;
    if (f16out) {
      if (relu) k_gemm_wmma<1, true,  true ><<<grid, 32, 0, stream>>>(A, lda, B, ldb, kp, Cf, Ch, ldc, jb);
      else      k_gemm_wmma<1, false, true ><<<grid, 32, 0, stream>>>(A, lda, B, ldb, kp, Cf, Ch, ldc, jb);
    } else {
      if (relu) k_gemm_wmma<1, true,  false><<<grid, 32, 0, stream>>>(A, lda, B, ldb, kp, Cf, Ch, ldc, jb);
      else      k_gemm_wmma<1, false, false><<<grid, 32, 0, stream>>>(A, lda, B, ldb, kp, Cf, Ch, ldc, jb);
    }
  }
}

// ---------------- launch ----------------

extern "C" void kernel_launch(void* const* d_in, const int* in_sizes, int n_in,
                              void* d_out, int out_size, void* d_ws, size_t ws_size,
                              hipStream_t stream) {
  (void)in_sizes; (void)n_in; (void)out_size; (void)ws_size;
  const int N = GN, E = GE, Q = GQ;

  const int*   u      = (const int*)d_in[0];
  const int*   v      = (const int*)d_in[1];
  const float* t      = (const float*)d_in[2];
  const int*   qu     = (const int*)d_in[3];
  const int*   qv     = (const int*)d_in[4];
  const float* h0     = (const float*)d_in[5];
  const float* w1_0   = (const float*)d_in[6];
  const float* w2_0   = (const float*)d_in[7];
  const float* w1_1   = (const float*)d_in[8];
  const float* w2_1   = (const float*)d_in[9];
  const float* link_h = (const float*)d_in[10];
  const float* link_o = (const float*)d_in[11];
  const float* node_h = (const float*)d_in[12];
  const float* node_o = (const float*)d_in[13];
  float* out = (float*)d_out;          // [0,Q): link ; [Q, Q+N*10): node

  char* ws = (char*)d_ws;
  size_t off = 0;
  auto alc = [&](size_t b) { size_t o = off; off += (b + 255) & ~(size_t)255; return o; };

  size_t o_g     = alc((size_t)E * 4);
  size_t o_agg0  = alc((size_t)N * 160 * 4);
  size_t o_agg1  = alc((size_t)N * 288 * 4);
  size_t o_h1    = alc((size_t)N * 272 * 4);
  size_t o_h2    = alc((size_t)N * 528 * 4);
  size_t o_h1cat = alc((size_t)N * 288 * 2);
  size_t o_a0h   = alc((size_t)N * 160 * 2);
  size_t o_h2cat = alc((size_t)N * 544 * 2);
  size_t o_stats = alc((size_t)4 * 288 * 4);
  size_t o_w10   = alc((size_t)144 * 160 * 2);
  size_t o_w20   = alc((size_t)272 * 288 * 2);
  size_t o_w11   = alc((size_t)272 * 288 * 2);
  size_t o_w21   = alc((size_t)528 * 544 * 2);
  size_t o_nh    = alc((size_t)128 * 544 * 2);
  size_t o_lh    = alc((size_t)128 * 1056 * 2);
  // phase-safe aliases
  size_t o_ynode = o_agg0;                                   // N*128*4 <= N*160*4
  size_t o_h2h   = o_agg1;                                   // N*544*2 <= N*288*4
  size_t o_a1h   = o_h1cat;                                  // same size
  size_t o_hq    = o_a0h;                                    // Q*1056*2 <= N*160*2
  size_t o_ylink = o_a0h + (((size_t)Q * 1056 * 2 + 255) & ~(size_t)255);

  float*    g_buf  = (float*)(ws + o_g);
  float*    agg0   = (float*)(ws + o_agg0);
  float*    agg1   = (float*)(ws + o_agg1);
  float*    h1     = (float*)(ws + o_h1);
  float*    h2     = (float*)(ws + o_h2);
  _Float16* h1cat  = (_Float16*)(ws + o_h1cat);
  _Float16* a0h    = (_Float16*)(ws + o_a0h);
  _Float16* a1h    = (_Float16*)(ws + o_a1h);
  _Float16* h2cat  = (_Float16*)(ws + o_h2cat);
  _Float16* h2h    = (_Float16*)(ws + o_h2h);
  _Float16* hqh    = (_Float16*)(ws + o_hq);
  float*    ynode  = (float*)(ws + o_ynode);
  float*    ylink  = (float*)(ws + o_ylink);
  float*    s_sum  = (float*)(ws + o_stats);
  float*    s_sq   = s_sum + 288;
  float*    s_mean = s_sum + 576;
  float*    s_inv  = s_sum + 864;
  _Float16* w10h   = (_Float16*)(ws + o_w10);
  _Float16* w20h   = (_Float16*)(ws + o_w20);
  _Float16* w11h   = (_Float16*)(ws + o_w11);
  _Float16* w21h   = (_Float16*)(ws + o_w21);
  _Float16* nhh    = (_Float16*)(ws + o_nh);
  _Float16* lhh    = (_Float16*)(ws + o_lh);

  auto cdiv = [](long long a, long long b) { return (int)((a + b - 1) / b); };

  // --- zero accumulators (fresh every call; graph-replay safe) ---
  hipMemsetAsync(agg0, 0, (size_t)N * 160 * 4, stream);
  hipMemsetAsync(agg1, 0, (size_t)N * 288 * 4, stream);

  // --- g and f16 weight pads ---
  k_g<<<cdiv(E, 256), 256, 0, stream>>>(t, g_buf, E);
  k_pad_f16<<<cdiv(144 * 160, 256), 256, 0, stream>>>(w1_0, 129, 129, w10h, 144, 160);
  k_pad_f16<<<cdiv(272 * 288, 256), 256, 0, stream>>>(w2_0, 257, 257, w20h, 272, 288);
  k_pad_f16<<<cdiv(272 * 288, 256), 256, 0, stream>>>(w1_1, 258, 258, w11h, 272, 288);
  k_pad_f16<<<cdiv(528 * 544, 256), 256, 0, stream>>>(w2_1, 515, 515, w21h, 528, 544);
  k_pad_f16<<<cdiv(128 * 544, 256), 256, 0, stream>>>(node_h, 128, 515, nhh, 128, 544);
  k_pad_f16<<<cdiv(128 * 1056, 256), 256, 0, stream>>>(link_h, 128, 1030, lhh, 128, 1056);

  // --- layer 0 ---
  // h1cat cols 0..127 = h0 (f16), rest zero (cols 128..271 overwritten by GEMM0 below)
  k_f32_to_f16_pad<<<cdiv((long long)N * 288, 256), 256, 0, stream>>>(h0, 128, 128, h1cat, 288, N);
  k_scatter<<<cdiv((long long)E * 32, 256), 256, 0, stream>>>(u, v, g_buf, h0, 128, 128, agg0, 160, E);
  hipMemsetAsync(s_sum, 0, 2 * 288 * 4, stream);
  k_bnstats<<<cdiv(N, 256), 160, 0, stream>>>(agg0, 160, N, s_sum, s_sq);
  k_bnfinal<<<1, 160, 0, stream>>>(s_sum, s_sq, s_mean, s_inv, N);
  k_bnnorm<<<cdiv((long long)N * 160, 256), 256, 0, stream>>>(agg0, s_mean, s_inv, a0h, 160, N);
  // z0 = relu(bn(agg0) @ w1_0^T) written as f16 into h1cat cols 128..256
  gemm_dispatch(a0h, 160, w10h, 160, 144, 160, nullptr, h1cat + 128, 288, true, N, stream);
  // h1 = [h0|z0] @ w2_0^T -> f32 (N,272; cols>=257 exact zeros via zero-padded B)
  gemm_dispatch(h1cat, 288, w20h, 288, 272, 288, h1, nullptr, 272, false, N, stream);

  // --- layer 1 ---
  // h2cat cols 0..256 = h1 (f16), rest zero (cols 257..528 overwritten by GEMM1 below)
  k_f32_to_f16_pad<<<cdiv((long long)N * 544, 256), 256, 0, stream>>>(h1, 272, 257, h2cat, 544, N);
  k_scatter<<<cdiv((long long)E * 32, 256), 256, 0, stream>>>(u, v, g_buf, h1, 272, 257, agg1, 288, E);
  hipMemsetAsync(s_sum, 0, 2 * 288 * 4, stream);
  k_bnstats<<<cdiv(N, 256), 288, 0, stream>>>(agg1, 288, N, s_sum, s_sq);
  k_bnfinal<<<1, 288, 0, stream>>>(s_sum, s_sq, s_mean, s_inv, N);
  k_bnnorm<<<cdiv((long long)N * 288, 256), 256, 0, stream>>>(agg1, s_mean, s_inv, a1h, 288, N);
  // z1 = relu(bn(agg1) @ w1_1^T) written as f16 into h2cat cols 257..514
  gemm_dispatch(a1h, 288, w11h, 288, 272, 288, nullptr, h2cat + 257, 544, true, N, stream);
  // h2 = [h1|z1] @ w2_1^T -> f32 (N,528)
  gemm_dispatch(h2cat, 544, w21h, 544, 528, 544, h2, nullptr, 528, false, N, stream);

  // --- node head ---
  k_f32_to_f16_pad<<<cdiv((long long)N * 544, 256), 256, 0, stream>>>(h2, 528, 515, h2h, 544, N);
  gemm_dispatch(h2h, 544, nhh, 544, 128, 544, ynode, nullptr, 128, true, N, stream);
  k_node_out<<<cdiv((long long)N * 10, 256), 256, 0, stream>>>(ynode, node_o, out + Q, N);

  // --- link head ---
  k_gather_hq<<<cdiv((long long)Q * 1056, 256), 256, 0, stream>>>(h2, qu, qv, hqh);
  gemm_dispatch(hqh, 1056, lhh, 1056, 128, 1056, ylink, nullptr, 128, true, Q, stream);
  k_link_out<<<cdiv(Q, 256), 256, 0, stream>>>(ylink, link_o, out);
}
